// Tacotron_40767829574432
// MI455X (gfx1250) — matrix operand order repaired and checked by reading, not compiled, not run on hardware
//
#include <hip/hip_runtime.h>
#include <hip/hip_bf16.h>
#include <cmath>

typedef __attribute__((ext_vector_type(16))) _Float16 v16h;
typedef __attribute__((ext_vector_type(8)))  _Float16 v8h;
typedef __attribute__((ext_vector_type(8)))  float    v8f;

#define DEV __device__ __forceinline__

#if __has_builtin(__builtin_amdgcn_sched_group_barrier)
#define SGB(mask, size, id) __builtin_amdgcn_sched_group_barrier((mask), (size), (id))
#else
#define SGB(mask, size, id)
#endif
// sched group masks: 0x002 VALU, 0x008 MFMA/WMMA, 0x020 VMEM read, 0x100 DS read

// ---------------- WMMA helpers (CDNA5, wave32) -------------------------------
// D(16x16,f32) = A(16x32,f16) x B(32x16,f16) + C
DEV v8f wmma16(v16h a, v16h b, v8f c) {
  return __builtin_amdgcn_wmma_f32_16x16x32_f16(false, a, false, b, (short)0, c,
                                                false, false);
}

DEV v16h pack16(float4 a0, float4 a1, float4 a2, float4 a3) {
  v16h r;
  r[0]=(_Float16)a0.x;  r[1]=(_Float16)a0.y;  r[2]=(_Float16)a0.z;  r[3]=(_Float16)a0.w;
  r[4]=(_Float16)a1.x;  r[5]=(_Float16)a1.y;  r[6]=(_Float16)a1.z;  r[7]=(_Float16)a1.w;
  r[8]=(_Float16)a2.x;  r[9]=(_Float16)a2.y;  r[10]=(_Float16)a2.z; r[11]=(_Float16)a2.w;
  r[12]=(_Float16)a3.x; r[13]=(_Float16)a3.y; r[14]=(_Float16)a3.z; r[15]=(_Float16)a3.w;
  return r;
}

// A fragment, row-major fp32 source. Lane layout (ISA 7.12.2, 16-bit A 16x32):
// lanes 0-15 hold K {0..7,16..23}, lanes 16-31 hold K {8..15,24..31}; the caller
// pre-offsets p by (lane>=16 ? 8 : 0) so elems are p[0..7] and p[16..23].
DEV v16h load_a16(const float* p) {
  return pack16(*(const float4*)(p),      *(const float4*)(p + 4),
                *(const float4*)(p + 16), *(const float4*)(p + 20));
}
// p[0..7], p[16..23] fp32 -> f16 (used for transposed-B reads)
DEV v16h load_ctg16(const float* p) {
  return pack16(*(const float4*)(p),      *(const float4*)(p + 4),
                *(const float4*)(p + 16), *(const float4*)(p + 20));
}
// f16 p[0..7], p[16..23] from LDS (two ds_load_b128)
DEV v16h lds_b16(const _Float16* p) {
  v8h lo = *(const v8h*)p;
  v8h hi = *(const v8h*)(p + 16);
  v16h r;
#pragma unroll
  for (int i = 0; i < 8; ++i) { r[i] = lo[i]; r[i + 8] = hi[i]; }
  return r;
}

DEV float sigm(float x) { return 1.0f / (1.0f + expf(-x)); }

// ---------------- Problem constants ------------------------------------------
// N=64 T=2048 E=512 A=256 D=256 O=80 S=64 H=1024 K=31 R=10 RF=2

// ---------------- small elementwise / MLP kernels ----------------------------

// pret[n][j] = relu([input_dec|spkr] @ pre_W1 + pre_b1)   (64 x 512)
__global__ void k_pre1(const float* __restrict__ idec, const float* __restrict__ spkr,
                       const float* __restrict__ W1, const float* __restrict__ b1,
                       float* __restrict__ pret) {
  int idx = blockIdx.x * 256 + threadIdx.x;           // 64*512
  int n = idx >> 9, j = idx & 511;
  float s = b1[j];
  for (int i = 0; i < 80; ++i) s += idec[n * 80 + i] * W1[i * 512 + j];
  for (int i = 0; i < 64; ++i) s += spkr[n * 64 + i] * W1[(80 + i) * 512 + j];
  pret[idx] = fmaxf(s, 0.0f);
}

// pre[n][d] = relu(pret @ pre_W2 + pre_b2)   (64 x 256)
__global__ void k_pre2(const float* __restrict__ pret, const float* __restrict__ W2,
                       const float* __restrict__ b2, float* __restrict__ pre) {
  int idx = blockIdx.x * 256 + threadIdx.x;           // 64*256
  int n = idx >> 8, d = idx & 255;
  float s = b2[d];
  for (int j = 0; j < 512; ++j) s += pret[n * 512 + j] * W2[j * 256 + d];
  pre[idx] = fmaxf(s, 0.0f);
}

// speed_add[n][e] = tanh(relu(speed*sp_W1+sp_b1) @ sp_W2 + sp_b2)  (64 x 512)
__global__ void k_speed_add(const float* __restrict__ speed, const float* __restrict__ W1,
                            const float* __restrict__ b1, const float* __restrict__ W2,
                            const float* __restrict__ b2, float* __restrict__ outp) {
  int idx = blockIdx.x * 256 + threadIdx.x;           // 64*512
  int n = idx >> 9, e = idx & 511;
  float sp = speed[n];
  float s = b2[e];
  for (int d = 0; d < 256; ++d) {
    float h = fmaxf(sp * W1[d] + b1[d], 0.0f);
    s += h * W2[d * 512 + e];
  }
  outp[idx] = tanhf(s);
}

// bias_na[n][a] += softsign(spkr @ W_spkr) + speed[n]*W_speed[a]
__global__ void k_bias_na(float* __restrict__ bias_na, const float* __restrict__ spkr,
                          const float* __restrict__ W_spkr, const float* __restrict__ speed,
                          const float* __restrict__ W_speed) {
  int idx = blockIdx.x * 256 + threadIdx.x;           // 64*256
  int n = idx >> 8, a = idx & 255;
  float s = 0.0f;
  for (int k = 0; k < 64; ++k) s += spkr[n * 64 + k] * W_spkr[k * 256 + a];
  s = s / (1.0f + fabsf(s));
  bias_na[idx] += s + speed[n] * W_speed[a];
}

// m[n] = argmax_t prev_att[n,t] (first occurrence)
__global__ void k_argmax(const float* __restrict__ prev, int* __restrict__ m) {
  int n = blockIdx.x, lane = threadIdx.x;
  float bv = -INFINITY; int bi = 0x7fffffff;
  for (int t = lane; t < 2048; t += 32) {
    float v = prev[n * 2048 + t];
    if (v > bv) { bv = v; bi = t; }
  }
  for (int off = 16; off > 0; off >>= 1) {
    float ov = __shfl_xor(bv, off, 32);
    int   oi = __shfl_xor(bi, off, 32);
    if (ov > bv || (ov == bv && oi < bi)) { bv = ov; bi = oi; }
  }
  if (lane == 0) m[n] = bi;
}

// masked windowed softmax + context = speed_add + sum_t w_t * input_enc[n,t,:]
__global__ __launch_bounds__(64) void k_context(
    const float* __restrict__ enc, const float* __restrict__ speed_add,
    const float* __restrict__ logit, const int* __restrict__ m,
    const int* __restrict__ lengths, float* __restrict__ ctx) {
  int n = blockIdx.x, tid = threadIdx.x;
  int Lm1 = lengths[n] - 1;
  int mm = m[n];
  int lo = min(max(mm - 9, 0), Lm1);
  int hi = min(mm + 9, Lm1);
  float mx = -INFINITY;
  for (int t = lo; t <= hi; ++t) mx = fmaxf(mx, logit[n * 2048 + t]);
  float s = 0.0f;
  for (int t = lo; t <= hi; ++t) s += expf(logit[n * 2048 + t] - mx);
  float inv = 1.0f / fmaxf(s, 1e-12f);
  for (int e = tid; e < 512; e += 64) {
    float acc = 0.0f;
    for (int t = lo; t <= hi; ++t)
      acc += expf(logit[n * 2048 + t] - mx) * enc[((size_t)n * 2048 + t) * 512 + e];
    ctx[n * 512 + e] = acc * inv + speed_add[n * 512 + e];
  }
}

// LSTM gate nonlinearity: h,c from pre-activation g (bias added here)
__global__ void k_gates(const float* __restrict__ g, const float* __restrict__ bih,
                        const float* __restrict__ bhh, const float* __restrict__ cprev,
                        float* __restrict__ hout, float* __restrict__ cout) {
  int idx = blockIdx.x * 256 + threadIdx.x;           // 64*1024
  int n = idx >> 10, j = idx & 1023;
  const float* gr = g + (size_t)n * 4096;
  float gi = gr[j]        + bih[j]        + bhh[j];
  float gf = gr[1024 + j] + bih[1024 + j] + bhh[1024 + j];
  float gg = gr[2048 + j] + bih[2048 + j] + bhh[2048 + j];
  float go = gr[3072 + j] + bih[3072 + j] + bhh[3072 + j];
  float c = sigm(gf) * cprev[idx] + sigm(gi) * tanhf(gg);
  hout[idx] = sigm(go) * tanhf(c);
  cout[idx] = c;
}

// ---------------- generic WMMA GEMM ------------------------------------------
// C[M,Nc] (+)= A[M,K] @ B ; one wave per 16x16 C tile; K % 32 == 0.
// BMODE 0: B row-major [K,Nc] (ldb = row stride)
// BMODE 1: B = W^T with W row-major [Nc,K] (ldb = W row stride) -> contiguous reads
template <int BMODE>
__global__ __launch_bounds__(32) void k_gemm(
    const float* __restrict__ A, const float* __restrict__ B,
    const float* __restrict__ bias, float* __restrict__ C,
    int Kd, int lda, int ldb, int ldc, int accum) {
  const int lane = threadIdx.x & 31;
  const int lm = lane & 15;
  const bool hih = lane >= 16;
  const int col = blockIdx.x * 16 + lm;
  const int arow = blockIdx.y * 16 + lm;
  const int klo = hih ? 8 : 0;
  const int kbB = hih ? 8 : 0;
  v8f acc = (v8f)0.0f;
  const float* ap = A + (size_t)arow * lda + klo;
  for (int k0 = 0; k0 < Kd; k0 += 32) {
    v16h Af = load_a16(ap + k0);
    v16h Bf;
    if (BMODE == 1) {
      Bf = load_ctg16(B + (size_t)col * ldb + k0 + kbB);
    } else {
      const float* bp = B + (size_t)(k0 + kbB) * ldb + col;
#pragma unroll
      for (int i = 0; i < 16; ++i) Bf[i] = (_Float16)bp[(size_t)(i < 8 ? i : i + 8) * ldb];
    }
    acc = wmma16(Af, Bf, acc);
  }
  const float bv = bias ? bias[col] : 0.0f;
  float* cp = C + (size_t)(blockIdx.y * 16 + (hih ? 8 : 0)) * ldc + col;
#pragma unroll
  for (int r = 0; r < 8; ++r) {
    float v = acc[r] + bv;
    if (accum) v += cp[(size_t)r * ldc];
    cp[(size_t)r * ldc] = v;
  }
}

// ---------------- fused attention-energy kernel ------------------------------
// logit[n,t] = tanh( softsign(enc@W_enc + b_enc) + bias_na + conv(prev) ) @ W_attproj + b
// Block: 256 rows (one n), 8 waves, each wave 2 row-tiles x 16 col-tiles.
// Full W_enc (f16, [a][k] padded) + conv_w (f16, K padded to 32) staged in LDS.
// Software pipeline enforced with sched_group_barrier:
//   VMEM(A prefetch) x8 | DS x6 | 13 x (MFMA x2, DS x2) | MFMA x6
constexpr int LDSB_STRIDE = 520;   // 512 k + pad (bank-spread, 16B aligned frags)
constexpr int LDSC_STRIDE = 40;    // 32 k + pad
constexpr size_t ESMEM =
    256 * LDSB_STRIDE * sizeof(_Float16) + 256 * LDSC_STRIDE * sizeof(_Float16) +
    288 * sizeof(float) + 3 * 256 * sizeof(float);

__global__ __launch_bounds__(256) void k_energy(
    const float* __restrict__ enc, const float* __restrict__ Wenc,
    const float* __restrict__ benc, const float* __restrict__ convw,
    const float* __restrict__ prev, const float* __restrict__ biasna,
    const float* __restrict__ wattp, const float* __restrict__ battp,
    float* __restrict__ logit) {
  extern __shared__ char smem[];
  _Float16* ldsB = (_Float16*)smem;                         // [256][520] f16
  _Float16* ldsC = ldsB + 256 * LDSB_STRIDE;                // [256][40]  f16
  float* ldsP  = (float*)(ldsC + 256 * LDSC_STRIDE);        // prev window 286
  float* ldsBE = ldsP + 288;                                // b_enc
  float* ldsWA = ldsBE + 256;                               // W_attproj
  float* ldsBN = ldsWA + 256;                               // bias_na[n]

  const int tid = threadIdx.x;
  const int n = blockIdx.x >> 3;
  const int t0 = (blockIdx.x & 7) * 256;

  for (int idx = tid; idx < 512 * 256; idx += 256) {        // W_enc -> LDS, [a][k]
    int k = idx >> 8, a = idx & 255;
    ldsB[a * LDSB_STRIDE + k] = (_Float16)Wenc[idx];
  }
  for (int idx = tid; idx < 256 * 32; idx += 256) {         // conv_w^T, K pad 31->32
    int a = idx >> 5, k = idx & 31;
    ldsC[a * LDSC_STRIDE + k] = (_Float16)((k < 31) ? convw[a * 31 + k] : 0.0f);
  }
  for (int i = tid; i < 286; i += 256) {                    // prev window, zero-pad
    int t = t0 - 15 + i;
    ldsP[i] = (t >= 0 && t < 2048) ? prev[n * 2048 + t] : 0.0f;
  }
  ldsBE[tid] = benc[tid];
  ldsWA[tid] = wattp[tid];
  ldsBN[tid] = biasna[n * 256 + tid];
  __syncthreads();

  const int wave = tid >> 5, lane = tid & 31;
  const int lm = lane & 15;
  const bool hih = lane >= 16;
  const int klo = hih ? 8 : 0;
  const int kbB = hih ? 8 : 0;

  // conv A fragments: row t = t0+tile*16+lm, element k -> prev[t-15+k]
  v16h convA[2];
#pragma unroll
  for (int rt = 0; rt < 2; ++rt) {
    const int tloc = (wave * 2 + rt) * 16 + lm;
    v16h f;
#pragma unroll
    for (int i = 0; i < 8; ++i) {
      int k1 = klo + i, k2 = klo + 16 + i;
      f[i]     = (_Float16)((k1 < 31) ? ldsP[tloc + k1] : 0.0f);
      f[i + 8] = (_Float16)((k2 < 31) ? ldsP[tloc + k2] : 0.0f);
    }
    convA[rt] = f;
  }

  v8f acc[2][16];
#pragma unroll
  for (int rt = 0; rt < 2; ++rt)
#pragma unroll
    for (int c = 0; c < 16; ++c) acc[rt][c] = (v8f)0.0f;

  const float* arow0 =
      enc + ((size_t)(n * 2048 + t0 + (wave * 2 + 0) * 16 + lm)) * 512 + klo;
  const float* arow1 =
      enc + ((size_t)(n * 2048 + t0 + (wave * 2 + 1) * 16 + lm)) * 512 + klo;
  const _Float16* ldsBrow = ldsB + (size_t)lm * LDSB_STRIDE + kbB;

  // A raw-data prefetch registers (one k-step ahead)
  float4 r00, r01, r02, r03, r10, r11, r12, r13;
  r00 = *(const float4*)(arow0 + 0);  r01 = *(const float4*)(arow0 + 4);
  r02 = *(const float4*)(arow0 + 16); r03 = *(const float4*)(arow0 + 20);
  r10 = *(const float4*)(arow1 + 0);  r11 = *(const float4*)(arow1 + 4);
  r12 = *(const float4*)(arow1 + 16); r13 = *(const float4*)(arow1 + 20);

  for (int k0 = 0; k0 < 512; k0 += 32) {
    v16h Af0 = pack16(r00, r01, r02, r03);
    v16h Af1 = pack16(r10, r11, r12, r13);
    // branchless next-k A prefetch (wraps to k=0 on last iter; keeps one BB so
    // sched_group_barrier governs the whole k-step)
    {
      const int kn = (k0 + 32) & 511;
      const float* p0 = arow0 + kn;
      const float* p1 = arow1 + kn;
      r00 = *(const float4*)(p0);      r01 = *(const float4*)(p0 + 4);
      r02 = *(const float4*)(p0 + 16); r03 = *(const float4*)(p0 + 20);
      r10 = *(const float4*)(p1);      r11 = *(const float4*)(p1 + 4);
      r12 = *(const float4*)(p1 + 16); r13 = *(const float4*)(p1 + 20);
    }
#pragma unroll
    for (int c = 0; c < 16; ++c) {
      v16h Bf = lds_b16(ldsBrow + (size_t)c * 16 * LDSB_STRIDE + k0);
      acc[0][c] = wmma16(Af0, Bf, acc[0][c]);
      acc[1][c] = wmma16(Af1, Bf, acc[1][c]);
    }
    // enforce pipelined issue order: A prefetch first, B loads run 3 tiles ahead
    SGB(0x020, 8, 0);                 // 8 global A loads
    SGB(0x100, 6, 0);                 // DS reads for col-tiles 0..2
#pragma unroll
    for (int i = 0; i < 13; ++i) {
      SGB(0x008, 2, 0);               // WMMA pair for tile i
      SGB(0x100, 2, 0);               // DS reads for tile i+3
    }
    SGB(0x008, 6, 0);                 // last three WMMA pairs
  }

  const float b0 = battp[0];
  const v8f vzero = (v8f)0.0f;
#pragma unroll
  for (int rt = 0; rt < 2; ++rt) {
    float ps[8];
#pragma unroll
    for (int r = 0; r < 8; ++r) ps[r] = 0.0f;
#pragma unroll
    for (int c = 0; c < 16; ++c) {
      const int a = c * 16 + lm;
      v16h Bc = lds_b16(ldsC + a * LDSC_STRIDE + kbB);
      v8f conv = wmma16(convA[rt], Bc, vzero);   // one WMMA = whole K=31 conv tile
      const float be = ldsBE[a], bn = ldsBN[a], wa = ldsWA[a];
#pragma unroll
      for (int r = 0; r < 8; ++r) {
        float x = acc[rt][c][r] + be;
        float e = x / (1.0f + fabsf(x)) + bn + conv[r];
        ps[r] += tanhf(e) * wa;
      }
    }
    // reduce over the 16 lanes of each half (columns a) -> logit rows
#pragma unroll
    for (int off = 1; off < 16; off <<= 1)
#pragma unroll
      for (int r = 0; r < 8; ++r) ps[r] += __shfl_xor(ps[r], off, 32);
    if (lm == 0) {
      const int tr = t0 + (wave * 2 + rt) * 16 + (hih ? 8 : 0);
#pragma unroll
      for (int r = 0; r < 8; ++r) logit[n * 2048 + tr + r] = ps[r] + b0;
    }
  }
}

// ---------------- host driver -------------------------------------------------
extern "C" void kernel_launch(void* const* d_in, const int* in_sizes, int n_in,
                              void* d_out, int out_size, void* d_ws, size_t ws_size,
                              hipStream_t stream) {
  (void)in_sizes; (void)n_in; (void)out_size; (void)ws_size;
  const float* input_enc = (const float*)d_in[0];
  const float* input_dec = (const float*)d_in[1];
  const float* spkr      = (const float*)d_in[2];
  const float* speed     = (const float*)d_in[3];
  const float* prev_att  = (const float*)d_in[4];
  const float* hidden    = (const float*)d_in[5];   // (2,N,H)
  const float* cell      = (const float*)d_in[6];   // (2,N,H)
  const int*   lengths   = (const int*)d_in[7];
  const float* W_enc     = (const float*)d_in[8];
  const float* b_enc     = (const float*)d_in[9];
  const float* W_spkr    = (const float*)d_in[10];
  const float* conv_w    = (const float*)d_in[11];
  const float* W_dec     = (const float*)d_in[12];
  const float* W_speed   = (const float*)d_in[13];
  const float* W_attproj = (const float*)d_in[14];
  const float* b_attproj = (const float*)d_in[15];
  const float* sp_W1     = (const float*)d_in[16];
  const float* sp_b1     = (const float*)d_in[17];
  const float* sp_W2     = (const float*)d_in[18];
  const float* sp_b2     = (const float*)d_in[19];
  const float* pre_W1    = (const float*)d_in[20];
  const float* pre_b1    = (const float*)d_in[21];
  const float* pre_W2    = (const float*)d_in[22];
  const float* pre_b2    = (const float*)d_in[23];
  const float* Wih0      = (const float*)d_in[24];  // (4096,832)
  const float* Whh0      = (const float*)d_in[25];  // (4096,1024)
  const float* bih0      = (const float*)d_in[26];
  const float* bhh0      = (const float*)d_in[27];
  const float* Wih1      = (const float*)d_in[28];
  const float* Whh1      = (const float*)d_in[29];
  const float* bih1      = (const float*)d_in[30];
  const float* bhh1      = (const float*)d_in[31];
  const float* Wo        = (const float*)d_in[32];  // (1536,160)
  const float* bo        = (const float*)d_in[33];

  float* out = (float*)d_out;          // (64,160) = (N,RF,O) flat
  float* ctx = out + 64 * 160;         // (64,512) second output (context)

  float* ws = (float*)d_ws;
  float* bias_na   = ws; ws += 64 * 256;
  float* speed_add = ws; ws += 64 * 512;
  float* pret      = ws; ws += 64 * 512;
  float* pre       = ws; ws += 64 * 256;
  float* logit     = ws; ws += 64 * 2048;
  float* g         = ws; ws += 64 * 4096;
  float* h0        = ws; ws += 64 * 1024;
  float* c0        = ws; ws += 64 * 1024;
  float* h1        = ws; ws += 64 * 1024;
  float* c1        = ws; ws += 64 * 1024;
  int*   m         = (int*)ws;

  // small precompute
  k_pre1<<<128, 256, 0, stream>>>(input_dec, spkr, pre_W1, pre_b1, pret);
  k_pre2<<<64, 256, 0, stream>>>(pret, pre_W2, pre_b2, pre);
  k_speed_add<<<128, 256, 0, stream>>>(speed, sp_W1, sp_b1, sp_W2, sp_b2, speed_add);
  k_argmax<<<64, 32, 0, stream>>>(prev_att, m);

  // in_att_dec = [h0|h1] @ W_dec  (split-K accumulate into bias_na), then + spkr/speed
  k_gemm<0><<<dim3(16, 4), 32, 0, stream>>>(hidden, W_dec, nullptr, bias_na,
                                            1024, 1024, 256, 256, 0);
  k_gemm<0><<<dim3(16, 4), 32, 0, stream>>>(hidden + 64 * 1024, W_dec + 1024 * 256,
                                            nullptr, bias_na, 1024, 1024, 256, 256, 1);
  k_bias_na<<<64, 256, 0, stream>>>(bias_na, spkr, W_spkr, speed, W_speed);

  // fused big GEMM + softsign + conv + tanh + attproj -> logit
  k_energy<<<512, 256, ESMEM, stream>>>(input_enc, W_enc, b_enc, conv_w, prev_att,
                                        bias_na, W_attproj, b_attproj, logit);

  // windowed softmax + context (writes second output directly)
  k_context<<<64, 64, 0, stream>>>(input_enc, speed_add, logit, m, lengths, ctx);

  // LSTM layer 0: g = [pre|ctx|spkr] @ Wih0^T + h_prev0 @ Whh0^T  (split-K accum)
  k_gemm<1><<<dim3(256, 4), 32, 0, stream>>>(pre,  Wih0,        nullptr, g, 256,  256,  832,  4096, 0);
  k_gemm<1><<<dim3(256, 4), 32, 0, stream>>>(ctx,  Wih0 + 256,  nullptr, g, 512,  512,  832,  4096, 1);
  k_gemm<1><<<dim3(256, 4), 32, 0, stream>>>(spkr, Wih0 + 768,  nullptr, g, 64,   64,   832,  4096, 1);
  k_gemm<1><<<dim3(256, 4), 32, 0, stream>>>(hidden, Whh0,      nullptr, g, 1024, 1024, 1024, 4096, 1);
  k_gates<<<256, 256, 0, stream>>>(g, bih0, bhh0, cell, h0, c0);

  // LSTM layer 1
  k_gemm<1><<<dim3(256, 4), 32, 0, stream>>>(h0, Wih1,              nullptr, g, 1024, 1024, 1024, 4096, 0);
  k_gemm<1><<<dim3(256, 4), 32, 0, stream>>>(hidden + 64 * 1024, Whh1, nullptr, g, 1024, 1024, 1024, 4096, 1);
  k_gates<<<256, 256, 0, stream>>>(g, bih1, bhh1, cell + 64 * 1024, h1, c1);

  // output = [h1|ctx] @ Wo + bo  (split-K accumulate) -> first output region
  k_gemm<0><<<dim3(10, 4), 32, 0, stream>>>(h1,  Wo,             bo,      out, 1024, 1024, 160, 160, 0);
  k_gemm<0><<<dim3(10, 4), 32, 0, stream>>>(ctx, Wo + 1024 * 160, nullptr, out, 512,  512,  160, 160, 1);
}
